// EdgeConvNet_64622077936100
// MI455X (gfx1250) — compile-verified, matrix operand order checked by reading, not run on hardware
//
#include <hip/hip_runtime.h>

// Problem constants (match reference)
#define Nn  50000
#define Ee  800000
#define Gg  64
#define INN 128
#define INE 16
#define Hh  64

typedef __attribute__((ext_vector_type(16))) _Float16 v16h;
typedef __attribute__((ext_vector_type(8)))  float    v8f;

#define WMMA16(a,b,c) __builtin_amdgcn_wmma_f32_16x16x32_f16(false,(a),false,(b),(short)0,(c),false,false)

// ---------------- fragment builders ----------------

// B fragment from weights pre-packed in fragment order: 2 x b128 per lane.
static __device__ inline v16h bfrag(const _Float16* __restrict__ Wp, int ksteps, int kstep, int ntile) {
  const uint4* base = (const uint4*)Wp + (((ntile * ksteps + kstep) * 32) + (threadIdx.x & 31)) * 2;
  union { uint4 q[2]; v16h h; } f;
  f.q[0] = base[0];
  f.q[1] = base[1];
  return f.h;
}

// A fragment (16x32 f16) from LDS, row-major with `stride` halves per row.
// ISA layout: per lane two contiguous 8-half chunks (K kb..kb+7 and kb+16..kb+23,
// shifted by 8 halves for lanes 16-31) -> 2 x ds_load_b128.
static __device__ inline v16h afrag_lds(const _Float16* __restrict__ wavebase, int stride, int kb) {
  int lane = threadIdx.x & 31;
  const _Float16* r = wavebase + (lane & 15) * stride + (lane >> 4) * 8 + kb;
  union { uint4 q[2]; v16h h; } f;
  f.q[0] = *(const uint4*)(r);
  f.q[1] = *(const uint4*)(r + 16);
  return f.h;
}

// A fragment built from a row of f32 in global memory (converted to f16), zero-padded past kvalid.
static __device__ inline v16h afrag_f32(const float* __restrict__ r, int kb, int kvalid) {
  int lane = threadIdx.x & 31;
  int lhi = lane >> 4;
  union { unsigned u[8]; v16h h; } f;
#pragma unroll
  for (int v = 0; v < 8; ++v) {
    int i = 2 * v;
    int k = kb + i + 8 * lhi + (i >= 8 ? 8 : 0);
    float a = 0.f, b = 0.f;
    if (k < kvalid) { float2 t = *(const float2*)(r + k); a = t.x; b = t.y; }
    union { _Float16 hh[2]; unsigned uu; } p;
    p.hh[0] = (_Float16)a; p.hh[1] = (_Float16)b;
    f.u[v] = p.uu;
  }
  return f.h;
}

// ---------------- weight pre-pack: f32 row-major [K x 64] -> f16 B-fragment order ----------------
__global__ void k_pack(const float* __restrict__ W, _Float16* __restrict__ o, int Kf, int Kv, int Nc) {
  int t = blockIdx.x * blockDim.x + threadIdx.x;
  int tot = Kf * Nc / 2;
  if (t >= tot) return;
  int v = t & 7;
  int lane = (t >> 3) & 31;
  int rest = t >> 8;
  int ksteps = Kf / 32;
  int kstep = rest % ksteps;
  int ntile = rest / ksteps;
  int n = (lane & 15) + ntile * 16;
  int lhi = lane >> 4;
  int i = 2 * v;
  int k = kstep * 32 + i + 8 * lhi + (i >= 8 ? 8 : 0);
  float a = (k < Kv) ? W[k * Nc + n] : 0.f;
  float b = (k + 1 < Kv) ? W[(k + 1) * Nc + n] : 0.f;
  union { _Float16 h[2]; unsigned u; } pk;
  pk.h[0] = (_Float16)a; pk.h[1] = (_Float16)b;
  ((unsigned*)o)[t] = pk.u;
}

// ---------------- node init: h = relu(x@W1+b1)@W2+b2 ----------------
__global__ __launch_bounds__(128) void k_init_node(
    const float* __restrict__ x,
    const _Float16* __restrict__ W1p, const float* __restrict__ b1,
    const _Float16* __restrict__ W2p, const float* __restrict__ b2,
    float* __restrict__ h32, _Float16* __restrict__ h16) {
  __shared__ __align__(16) _Float16 mid[4 * 16 * 72];
  int wave = threadIdx.x >> 5, lane = threadIdx.x & 31;
  int lhi = lane >> 4;
  int rbase = (blockIdx.x * 4 + wave) * 16;
  int row = rbase + (lane & 15);
  if (row >= Nn) row = Nn - 1;
  const float* xr = x + (size_t)row * INN;

  v8f acc[4] = {};
#pragma unroll
  for (int ks = 0; ks < INN / 32; ++ks) {
    v16h a = afrag_f32(xr, ks * 32, INN);
#pragma unroll
    for (int nt = 0; nt < 4; ++nt) acc[nt] = WMMA16(a, bfrag(W1p, INN / 32, ks, nt), acc[nt]);
  }
  _Float16* mrow = mid + wave * (16 * 72);
#pragma unroll
  for (int nt = 0; nt < 4; ++nt) {
    int n = (lane & 15) + nt * 16;
    float bb = b1[n];
#pragma unroll
    for (int v = 0; v < 8; ++v) {
      float val = acc[nt][v] + bb;
      mrow[(v + 8 * lhi) * 72 + n] = (_Float16)(val > 0.f ? val : 0.f);
    }
  }
  v8f acc2[4] = {};
#pragma unroll
  for (int ks = 0; ks < 2; ++ks) {
    v16h a = afrag_lds(mrow, 72, ks * 32);
#pragma unroll
    for (int nt = 0; nt < 4; ++nt) acc2[nt] = WMMA16(a, bfrag(W2p, 2, ks, nt), acc2[nt]);
  }
#pragma unroll
  for (int nt = 0; nt < 4; ++nt) {
    int n = (lane & 15) + nt * 16;
    float bb = b2[n];
#pragma unroll
    for (int v = 0; v < 8; ++v) {
      int r = rbase + v + 8 * lhi;
      if (r < Nn) {
        float val = acc2[nt][v] + bb;
        h32[(size_t)r * Hh + n] = val;
        h16[(size_t)r * Hh + n] = (_Float16)val;
      }
    }
  }
}

// ---------------- edge init: ea = relu(edge_attr@W1+b1)@W2+b2 (K=16 zero-padded to 32) ----------------
__global__ __launch_bounds__(128) void k_init_edge(
    const float* __restrict__ eattr,
    const _Float16* __restrict__ W1p, const float* __restrict__ b1,
    const _Float16* __restrict__ W2p, const float* __restrict__ b2,
    _Float16* __restrict__ ea16) {
  __shared__ __align__(16) _Float16 mid[4 * 16 * 72];
  int wave = threadIdx.x >> 5, lane = threadIdx.x & 31;
  int lhi = lane >> 4;
  int ebase = (blockIdx.x * 4 + wave) * 16;
  int e = ebase + (lane & 15);
  if (e >= Ee) e = Ee - 1;
  const float* r = eattr + (size_t)e * INE;

  v8f acc[4] = {};
  {
    v16h a = afrag_f32(r, 0, INE);
#pragma unroll
    for (int nt = 0; nt < 4; ++nt) acc[nt] = WMMA16(a, bfrag(W1p, 1, 0, nt), acc[nt]);
  }
  _Float16* mrow = mid + wave * (16 * 72);
#pragma unroll
  for (int nt = 0; nt < 4; ++nt) {
    int n = (lane & 15) + nt * 16;
    float bb = b1[n];
#pragma unroll
    for (int v = 0; v < 8; ++v) {
      float val = acc[nt][v] + bb;
      mrow[(v + 8 * lhi) * 72 + n] = (_Float16)(val > 0.f ? val : 0.f);
    }
  }
  v8f acc2[4] = {};
#pragma unroll
  for (int ks = 0; ks < 2; ++ks) {
    v16h a = afrag_lds(mrow, 72, ks * 32);
#pragma unroll
    for (int nt = 0; nt < 4; ++nt) acc2[nt] = WMMA16(a, bfrag(W2p, 2, ks, nt), acc2[nt]);
  }
#pragma unroll
  for (int nt = 0; nt < 4; ++nt) {
    int n = (lane & 15) + nt * 16;
    float bb = b2[n];
#pragma unroll
    for (int v = 0; v < 8; ++v) {
      int em = ebase + v + 8 * lhi;
      if (em < Ee) ea16[(size_t)em * Hh + n] = (_Float16)(acc2[nt][v] + bb);
    }
  }
}

// ---------------- conv layer: per-edge MLP([h_dst,h_src,ea]) scatter-added to agg[dst] ----------------
// Async gather: each wave gathers 16 edges x 192 halves into LDS via
// global_load_async_to_lds_b128 (ASYNCcnt), then runs 24+8 WMMAs, then f32 atomic scatter.
__global__ __launch_bounds__(128) void k_conv(
    const _Float16* __restrict__ h16, const _Float16* __restrict__ ea16,
    const int* __restrict__ ei,
    const _Float16* __restrict__ W1p, const float* __restrict__ b1,
    const _Float16* __restrict__ W2p, const float* __restrict__ b2,
    float* __restrict__ agg) {
  __shared__ __align__(16) _Float16 feat[4 * 16 * 200]; // 200-half row stride (bank-conflict pad)
  __shared__ __align__(16) _Float16 mid[4 * 16 * 72];
  int wave = threadIdx.x >> 5, lane = threadIdx.x & 31;
  int lhi = lane >> 4;
  int ebase = (blockIdx.x * 4 + wave) * 16;
  int e = ebase + (lane & 15);
  if (e >= Ee) e = Ee - 1;
  int my_src = ei[e];       // edge_index[0] = source (x_j)
  int my_dst = ei[Ee + e];  // edge_index[1] = target (x_i)

  _Float16* frow = feat + wave * (16 * 200);
  unsigned lbase = (unsigned)(unsigned long long)frow; // low 32 bits of LDS generic addr = LDS offset

  // 16 rows x 128B per source, 8 x b128 chunks per row -> 128 chunks over 32 lanes x 4 iters
#pragma unroll
  for (int j = 0; j < 4; ++j) {
    int lin = lane + 32 * j;
    int row = lin >> 3, ch = lin & 7;
    int d = __shfl(my_dst, row);
    int s = __shfl(my_src, row);
    int er = ebase + row; if (er >= Ee) er = Ee - 1;
    unsigned ld = lbase + (unsigned)(row * 400) + (unsigned)(ch * 16);
    const char* gd = (const char*)h16 + (size_t)d * 128 + ch * 16;
    const char* gs = (const char*)h16 + (size_t)s * 128 + ch * 16;
    const char* ge = (const char*)ea16 + (size_t)er * 128 + ch * 16;
    asm volatile("global_load_async_to_lds_b128 %0, %1, off" :: "v"(ld),        "v"(gd) : "memory");
    asm volatile("global_load_async_to_lds_b128 %0, %1, off" :: "v"(ld + 128u), "v"(gs) : "memory");
    asm volatile("global_load_async_to_lds_b128 %0, %1, off" :: "v"(ld + 256u), "v"(ge) : "memory");
  }
  asm volatile("s_wait_asynccnt 0" ::: "memory");

  // GEMM1: [16x192] x [192x64], 6 k-steps x 4 n-tiles
  v8f acc[4] = {};
#pragma unroll
  for (int ks = 0; ks < 6; ++ks) {
    v16h a = afrag_lds(frow, 200, ks * 32);
#pragma unroll
    for (int nt = 0; nt < 4; ++nt) acc[nt] = WMMA16(a, bfrag(W1p, 6, ks, nt), acc[nt]);
  }
  // bias + relu, C-layout -> A-layout relayout through LDS
  _Float16* mrow = mid + wave * (16 * 72);
#pragma unroll
  for (int nt = 0; nt < 4; ++nt) {
    int n = (lane & 15) + nt * 16;
    float bb = b1[n];
#pragma unroll
    for (int v = 0; v < 8; ++v) {
      float val = acc[nt][v] + bb;
      mrow[(v + 8 * lhi) * 72 + n] = (_Float16)(val > 0.f ? val : 0.f);
    }
  }
  // GEMM2: [16x64] x [64x64]
  v8f acc2[4] = {};
#pragma unroll
  for (int ks = 0; ks < 2; ++ks) {
    v16h a = afrag_lds(mrow, 72, ks * 32);
#pragma unroll
    for (int nt = 0; nt < 4; ++nt) acc2[nt] = WMMA16(a, bfrag(W2p, 2, ks, nt), acc2[nt]);
  }
  // scatter-add messages to agg[dst]
  int drow[8];
#pragma unroll
  for (int v = 0; v < 8; ++v) drow[v] = __shfl(my_dst, v + 8 * lhi);
#pragma unroll
  for (int nt = 0; nt < 4; ++nt) {
    int n = (lane & 15) + nt * 16;
    float bb = b2[n];
#pragma unroll
    for (int v = 0; v < 8; ++v) {
      int em = ebase + v + 8 * lhi;
      if (em < Ee) unsafeAtomicAdd(&agg[(size_t)drow[v] * Hh + n], acc2[nt][v] + bb);
    }
  }
}

// ---------------- residual + relu, refresh f16 mirror ----------------
__global__ void k_residual(float* __restrict__ h32, const float* __restrict__ agg,
                           _Float16* __restrict__ h16, int n) {
  int i = blockIdx.x * blockDim.x + threadIdx.x;
  if (i < n) {
    float v = h32[i] + agg[i];
    v = v > 0.f ? v : 0.f;
    h32[i] = v;
    h16[i] = (_Float16)v;
  }
}

// ---------------- global_add_pool over batch ----------------
__global__ void k_pool(const float* __restrict__ nodeout, const int* __restrict__ batch,
                       float* __restrict__ out) {
  int i = blockIdx.x * blockDim.x + threadIdx.x;
  if (i < Nn * Hh) {
    int vtx = i >> 6, c = i & 63;
    unsafeAtomicAdd(&out[batch[vtx] * Hh + c], nodeout[i]);
  }
}

// ---------------- launch ----------------
extern "C" void kernel_launch(void* const* d_in, const int* in_sizes, int n_in,
                              void* d_out, int out_size, void* d_ws, size_t ws_size,
                              hipStream_t stream) {
  (void)in_sizes; (void)n_in; (void)out_size; (void)ws_size;
  const float* x    = (const float*)d_in[0];
  const float* eatt = (const float*)d_in[1];
  const float* nW1  = (const float*)d_in[2];
  const float* nb1  = (const float*)d_in[3];
  const float* nW2  = (const float*)d_in[4];
  const float* nb2  = (const float*)d_in[5];
  const float* eW1  = (const float*)d_in[6];
  const float* eb1  = (const float*)d_in[7];
  const float* eW2  = (const float*)d_in[8];
  const float* eb2  = (const float*)d_in[9];
  const float* cW1[3] = {(const float*)d_in[10], (const float*)d_in[14], (const float*)d_in[18]};
  const float* cb1[3] = {(const float*)d_in[11], (const float*)d_in[15], (const float*)d_in[19]};
  const float* cW2[3] = {(const float*)d_in[12], (const float*)d_in[16], (const float*)d_in[20]};
  const float* cb2[3] = {(const float*)d_in[13], (const float*)d_in[17], (const float*)d_in[21]};
  const int* ei    = (const int*)d_in[22];
  const int* batch = (const int*)d_in[23];

  char* p = (char*)d_ws;
  auto alloc = [&](size_t bytes) -> char* {
    char* r = p; p += (bytes + 255) & ~(size_t)255; return r;
  };
  float*    h32  = (float*)alloc((size_t)Nn * Hh * 4);
  float*    agg  = (float*)alloc((size_t)Nn * Hh * 4);
  _Float16* h16  = (_Float16*)alloc((size_t)Nn * Hh * 2);
  _Float16* ea16 = (_Float16*)alloc((size_t)Ee * Hh * 2);
  _Float16* nW1p = (_Float16*)alloc(128 * 64 * 2);
  _Float16* nW2p = (_Float16*)alloc(64 * 64 * 2);
  _Float16* eW1p = (_Float16*)alloc(32 * 64 * 2);   // K padded 16 -> 32
  _Float16* eW2p = (_Float16*)alloc(64 * 64 * 2);
  _Float16* cW1p[3]; _Float16* cW2p[3];
  for (int l = 0; l < 3; ++l) {
    cW1p[l] = (_Float16*)alloc(192 * 64 * 2);
    cW2p[l] = (_Float16*)alloc(64 * 64 * 2);
  }

  auto pack = [&](const float* W, _Float16* o, int Kf, int Kv) {
    int tot = Kf * 64 / 2;
    k_pack<<<(tot + 255) / 256, 256, 0, stream>>>(W, o, Kf, Kv, 64);
  };
  pack(nW1, nW1p, 128, 128);
  pack(nW2, nW2p, 64, 64);
  pack(eW1, eW1p, 32, 16);
  pack(eW2, eW2p, 64, 64);
  for (int l = 0; l < 3; ++l) { pack(cW1[l], cW1p[l], 192, 192); pack(cW2[l], cW2p[l], 64, 64); }

  int nodeBlocks = ((Nn + 15) / 16 + 3) / 4;        // 782
  k_init_node<<<nodeBlocks, 128, 0, stream>>>(x, nW1p, nb1, nW2p, nb2, h32, h16);
  k_init_edge<<<Ee / 64, 128, 0, stream>>>(eatt, eW1p, eb1, eW2p, eb2, ea16);

  for (int l = 0; l < 3; ++l) {
    hipMemsetAsync(agg, 0, (size_t)Nn * Hh * 4, stream);
    k_conv<<<Ee / 64, 128, 0, stream>>>(h16, ea16, ei, cW1p[l], cb1[l], cW2p[l], cb2[l], agg);
    if (l < 2)
      k_residual<<<(Nn * Hh + 255) / 256, 256, 0, stream>>>(h32, agg, h16, Nn * Hh);
  }
  hipMemsetAsync(d_out, 0, (size_t)Gg * Hh * 4, stream);
  k_pool<<<(Nn * Hh + 255) / 256, 256, 0, stream>>>(agg, batch, (float*)d_out);
}
